// Head_8916352107081
// MI455X (gfx1250) — matrix operand order repaired and checked by reading, not compiled, not run on hardware
//
#include <hip/hip_runtime.h>
#include <hip/hip_bf16.h>

// MI455X (gfx1250) single-head causal attention, wave32 + v_wmma_f32_16x16x32_bf16.
// B=4, T=4096, C=1024, H=64.
// Workspace layout: x_bf16 (32MB) | W_bf16 k,q,v (384KB) | q,k,v bf16 (6MB)

#define N_EMBD 1024
#define HEAD   64
#define TSEQ   4096
#define BATCH  4

typedef __attribute__((ext_vector_type(16))) __bf16 v16bf;
typedef __attribute__((ext_vector_type(8)))  __bf16 v8bf;
typedef __attribute__((ext_vector_type(8)))  float  v8f;

// ---------------------------------------------------------------------------
// Kernel 0: elementwise f32 -> bf16 (native packed conversion, RNE).
// 8 elements per thread, fully coalesced (32B loads / 16B stores).
// ---------------------------------------------------------------------------
__global__ __launch_bounds__(256)
void cvt_bf16_kernel(const float* __restrict__ src, __bf16* __restrict__ dst) {
  const size_t i = ((size_t)blockIdx.x * 256 + threadIdx.x) * 8;
  v8f f = *(const v8f*)(src + i);
  v8bf b = __builtin_convertvector(f, v8bf);
  *(v8bf*)(dst + i) = b;
}

// ---------------------------------------------------------------------------
// Kernel 1: q/k/v projections.  GEMM M=B*T=16384, N=64, K=1024 (x * W^T),
// all-bf16 operands (pre-converted), f32 accumulation.
// 128 threads = 4 waves; wave w computes 16 rows x 64 cols for all 3 weights.
// The A fragment (x chunk, 16x32) is shared across the 3 weight matrices:
// 12 WMMAs per k-step per wave with only 14 vector loads feeding them.
// q is pre-scaled by C^-0.5 = 0.03125 so attention only needs exp.
// ---------------------------------------------------------------------------
__global__ __launch_bounds__(128)
void proj_qkv_kernel(const __bf16* __restrict__ xb,
                     const __bf16* __restrict__ wb,   // [3][64][1024] (k,q,v)
                     __bf16* __restrict__ qb,
                     __bf16* __restrict__ kb,
                     __bf16* __restrict__ vb) {
  const int tid  = threadIdx.x;
  const int wave = tid >> 5;
  const int lane = tid & 31;
  const int g    = lane >> 4;   // lane half (0/1)
  const int li   = lane & 15;
  const int m0   = blockIdx.x * 64 + wave * 16;

  v8f acc[3][4];
  #pragma unroll
  for (int m = 0; m < 3; ++m)
    #pragma unroll
    for (int n = 0; n < 4; ++n) acc[m][n] = (v8f)0.0f;

  const __bf16* xrow = xb + (size_t)(m0 + li) * N_EMBD;  // A layout: row = lane%16

  for (int kk = 0; kk < N_EMBD / 32; ++kk) {
    // A fragment 16x32: element e -> K = kk*32 + 16g + e
    v8bf lo = *(const v8bf*)(xrow + kk * 32 + 16 * g);
    v8bf hi = *(const v8bf*)(xrow + kk * 32 + 16 * g + 8);
    v16bf a = __builtin_shufflevector(lo, hi,
        0, 1, 2, 3, 4, 5, 6, 7, 8, 9, 10, 11, 12, 13, 14, 15);
    #pragma unroll
    for (int m = 0; m < 3; ++m) {
      #pragma unroll
      for (int nt = 0; nt < 4; ++nt) {
        // B fragment 32x16: N = nt*16+lane%16 (W row h), K = kk*32+16g+e (32B contig)
        const __bf16* wp =
            wb + (size_t)m * 64 * N_EMBD + (size_t)(nt * 16 + li) * N_EMBD
               + kk * 32 + 16 * g;
        v16bf bf = *(const v16bf*)wp;
        acc[m][nt] = __builtin_amdgcn_wmma_f32_16x16x32_bf16(
            false, a, false, bf, (short)0, acc[m][nt], false, false);
      }
    }
  }

  __bf16* dsts[3] = {kb, qb, vb};
  #pragma unroll
  for (int m = 0; m < 3; ++m) {
    const float sc = (m == 1) ? 0.03125f : 1.0f;  // fold C^-0.5 into q
    #pragma unroll
    for (int nt = 0; nt < 4; ++nt) {
      #pragma unroll
      for (int r = 0; r < 8; ++r) {
        // C/D layout: M = r + 8g, N = lane%16
        int row = m0 + r + 8 * g;
        int col = nt * 16 + li;
        dsts[m][(size_t)row * HEAD + col] = static_cast<__bf16>(acc[m][nt][r] * sc);
      }
    }
  }
}

// ---------------------------------------------------------------------------
// Kernel 2: causal flash attention. 256 threads = 8 waves; q-block = 128 rows
// (16 per wave), key blocks of 64 streamed through LDS.  V stored transposed
// in LDS so the P*V B-fragments are contiguous reads.  Online softmax in f32.
// ---------------------------------------------------------------------------
__global__ __launch_bounds__(256)
void flash_attn_kernel(const __bf16* __restrict__ qb,
                       const __bf16* __restrict__ kb,
                       const __bf16* __restrict__ vb,
                       float* __restrict__ out) {
  __shared__ alignas(32) __bf16 Kt[64 * 64];        // [s][c]
  __shared__ alignas(32) __bf16 Vt[64 * 64];        // transposed [h][s]
  __shared__ alignas(32) __bf16 Pst[8][16 * 64];    // per-wave P staging [m][s]

  const int tid  = threadIdx.x;
  const int wave = tid >> 5;
  const int lane = tid & 31;
  const int g    = lane >> 4;
  const int li   = lane & 15;

  const int bIdx = blockIdx.x >> 5;       // 32 q-blocks per batch (4096/128)
  const int q0   = (blockIdx.x & 31) * 128;
  const int qbase = q0 + wave * 16;

  const __bf16* qptr = qb + (size_t)bIdx * TSEQ * HEAD;
  const __bf16* kptr = kb + (size_t)bIdx * TSEQ * HEAD;
  const __bf16* vptr = vb + (size_t)bIdx * TSEQ * HEAD;

  // Q fragments (A layout, K=64 -> 2 fragments), resident for whole kernel.
  v16bf qfrag[2];
  {
    const __bf16* qrow = qptr + (size_t)(qbase + li) * HEAD;
    #pragma unroll
    for (int ks = 0; ks < 2; ++ks) {
      v8bf lo = *(const v8bf*)(qrow + 16 * g + 32 * ks);
      v8bf hi = *(const v8bf*)(qrow + 16 * g + 32 * ks + 8);
      qfrag[ks] = __builtin_shufflevector(lo, hi,
          0, 1, 2, 3, 4, 5, 6, 7, 8, 9, 10, 11, 12, 13, 14, 15);
    }
  }

  v8f Oacc[4];
  #pragma unroll
  for (int ht = 0; ht < 4; ++ht) Oacc[ht] = (v8f)0.0f;
  float mrow[8], lrow[8];
  #pragma unroll
  for (int r = 0; r < 8; ++r) { mrow[r] = -__builtin_inff(); lrow[r] = 0.0f; }

  const float LOG2E = 1.4426950408889634f;
  const int jmax = (q0 + 127) >> 6;          // inclusive key-block bound
  const int wave_max_row = qbase + 15;

  for (int j = 0; j <= jmax; ++j) {
    __syncthreads();   // protect LDS tiles from previous iteration's readers
    // K tile: 64 rows x 64 bf16 = 8KB, contiguous in global -> uint4 copies
    {
      const uint4* src = (const uint4*)(kptr + (size_t)(j * 64) * HEAD);
      uint4* dst = (uint4*)Kt;
      dst[tid]       = src[tid];
      dst[tid + 256] = src[tid + 256];
    }
    // V tile transposed: Vt[h][s] = v[s][h]
    {
      const int hp = tid & 31;    // h pair
      const int sb = tid >> 5;
      #pragma unroll
      for (int pass = 0; pass < 8; ++pass) {
        int s = pass * 8 + sb;
        unsigned w = *(const unsigned*)(vptr + (size_t)(j * 64 + s) * HEAD + 2 * hp);
        Vt[(2 * hp)     * 64 + s] = __builtin_bit_cast(__bf16, (unsigned short)(w & 0xFFFFu));
        Vt[(2 * hp + 1) * 64 + s] = __builtin_bit_cast(__bf16, (unsigned short)(w >> 16));
      }
    }
    // Prefetch next iteration's K/V tiles toward L2 while we compute.
    if (j < jmax) {
      __builtin_prefetch(kptr + (size_t)(j + 1) * 64 * HEAD + tid * 16, 0, 1);
      __builtin_prefetch(vptr + (size_t)(j + 1) * 64 * HEAD + tid * 16, 0, 1);
    }
    __syncthreads();

    if (j * 64 > wave_max_row) continue;  // fully-masked block: barriers only

    // S = Q * K^T   (4 s-subtiles x 2 k-steps = 8 WMMAs)
    v8f Sacc[4];
    #pragma unroll
    for (int st = 0; st < 4; ++st) {
      v8f s = (v8f)0.0f;
      #pragma unroll
      for (int ks = 0; ks < 2; ++ks) {
        v16bf bf = *(const v16bf*)&Kt[(st * 16 + li) * 64 + 32 * ks + 16 * g];
        s = __builtin_amdgcn_wmma_f32_16x16x32_bf16(
            false, qfrag[ks], false, bf, (short)0, s, false, false);
      }
      Sacc[st] = s;
    }

    // causal mask + online softmax (rows r+8g, cols li; reduce across 16 lanes)
    float newm[8], alpha[8];
    #pragma unroll
    for (int r = 0; r < 8; ++r) {
      const int mg = qbase + r + 8 * g;
      float mx = -__builtin_inff();
      #pragma unroll
      for (int st = 0; st < 4; ++st) {
        int sg = j * 64 + st * 16 + li;
        float vv = (sg > mg) ? -__builtin_inff() : Sacc[st][r];
        Sacc[st][r] = vv;
        mx = fmaxf(mx, vv);
      }
      mx = fmaxf(mx, __shfl_xor(mx, 1));
      mx = fmaxf(mx, __shfl_xor(mx, 2));
      mx = fmaxf(mx, __shfl_xor(mx, 4));
      mx = fmaxf(mx, __shfl_xor(mx, 8));
      float nm = fmaxf(mrow[r], mx);       // nm finite: diagonal always present
      newm[r]  = nm;
      alpha[r] = __builtin_exp2f((mrow[r] - nm) * LOG2E);
      mrow[r]  = nm;
    }
    #pragma unroll
    for (int r = 0; r < 8; ++r) {
      float sum = 0.0f;
      #pragma unroll
      for (int st = 0; st < 4; ++st) {
        float p = __builtin_exp2f((Sacc[st][r] - newm[r]) * LOG2E);
        Sacc[st][r] = p;
        sum += p;
      }
      sum += __shfl_xor(sum, 1);
      sum += __shfl_xor(sum, 2);
      sum += __shfl_xor(sum, 4);
      sum += __shfl_xor(sum, 8);
      lrow[r] = lrow[r] * alpha[r] + sum;
    }

    // P: C/D layout -> A layout via per-wave LDS staging (DS ops are in-order
    // within a wave; backend inserts s_wait_dscnt on the dependent reads)
    __bf16* pw = &Pst[wave][0];
    #pragma unroll
    for (int st = 0; st < 4; ++st)
      #pragma unroll
      for (int r = 0; r < 8; ++r)
        pw[(r + 8 * g) * 64 + st * 16 + li] = static_cast<__bf16>(Sacc[st][r]);

    #pragma unroll
    for (int ht = 0; ht < 4; ++ht)
      #pragma unroll
      for (int r = 0; r < 8; ++r)
        Oacc[ht][r] *= alpha[r];

    // O += P * V   (2 k-steps x 4 h-subtiles = 8 WMMAs)
    #pragma unroll
    for (int ks = 0; ks < 2; ++ks) {
      v16bf pf;
      {
        const __bf16* prow = pw + li * 64;
        v8bf lo = *(const v8bf*)(prow + 16 * g + 32 * ks);
        v8bf hi = *(const v8bf*)(prow + 16 * g + 32 * ks + 8);
        pf = __builtin_shufflevector(lo, hi,
            0, 1, 2, 3, 4, 5, 6, 7, 8, 9, 10, 11, 12, 13, 14, 15);
      }
      #pragma unroll
      for (int ht = 0; ht < 4; ++ht) {
        v16bf vf = *(const v16bf*)&Vt[(ht * 16 + li) * 64 + 32 * ks + 16 * g];
        Oacc[ht] = __builtin_amdgcn_wmma_f32_16x16x32_bf16(
            false, pf, false, vf, (short)0, Oacc[ht], false, false);
      }
    }
  }

  // out = O / l  (f32)
  float* orow = out + (size_t)bIdx * TSEQ * HEAD;
  #pragma unroll
  for (int r = 0; r < 8; ++r) {
    float inv = 1.0f / lrow[r];
    int row = qbase + r + 8 * g;
    #pragma unroll
    for (int ht = 0; ht < 4; ++ht)
      orow[(size_t)row * HEAD + ht * 16 + li] = Oacc[ht][r] * inv;
  }
}

// ---------------------------------------------------------------------------
extern "C" void kernel_launch(void* const* d_in, const int* in_sizes, int n_in,
                              void* d_out, int out_size, void* d_ws, size_t ws_size,
                              hipStream_t stream) {
  (void)in_sizes; (void)n_in; (void)out_size; (void)ws_size;
  const float* x  = (const float*)d_in[0];
  const float* Wk = (const float*)d_in[1];
  const float* Wq = (const float*)d_in[2];
  const float* Wv = (const float*)d_in[3];

  const size_t nx = (size_t)BATCH * TSEQ * N_EMBD;   // 16M elements
  const size_t nw = (size_t)HEAD * N_EMBD;           // 64K elements per W

  __bf16* xb   = (__bf16*)d_ws;                      // 32 MB
  __bf16* wb   = xb + nx;                            // 3 * 128 KB (k,q,v)
  __bf16* qbuf = wb + 3 * nw;                        // 2 MB
  __bf16* kbuf = qbuf + (size_t)BATCH * TSEQ * HEAD; // 2 MB
  __bf16* vbuf = kbuf + (size_t)BATCH * TSEQ * HEAD; // 2 MB

  // f32 -> bf16 pre-conversion (one pass; removes all conversion VALU from GEMMs)
  cvt_bf16_kernel<<<nx / (256 * 8), 256, 0, stream>>>(x,  xb);
  cvt_bf16_kernel<<<nw / (256 * 8), 256, 0, stream>>>(Wk, wb);
  cvt_bf16_kernel<<<nw / (256 * 8), 256, 0, stream>>>(Wq, wb + nw);
  cvt_bf16_kernel<<<nw / (256 * 8), 256, 0, stream>>>(Wv, wb + 2 * nw);

  proj_qkv_kernel<<<(BATCH * TSEQ) / 64, 128, 0, stream>>>(
      xb, wb, qbuf, kbuf, vbuf);
  flash_attn_kernel<<<(BATCH * TSEQ) / 128, 256, 0, stream>>>(
      qbuf, kbuf, vbuf, (float*)d_out);
}
